// REClipper_979252543760
// MI455X (gfx1250) — matrix operand order repaired and checked by reading, not compiled, not run on hardware
//
#include <hip/hip_runtime.h>

// ---------------------------------------------------------------------------
// REClipper pipeline for MI455X (gfx1250, wave32).
//
//   enc_lif_conv_kernel : fused LIF scan (64 steps) + 3x3 conv(1->4) + relu +
//                         3x3 conv(4->1) + relu + temporal mean. VALU + LDS.
//   clip1_wmma_kernel   : conv3x3 (5->16) + bias + relu, implicit GEMM,
//                         K=45->64, v_wmma_f32_16x16x32_f16.
//   clip2_wmma_kernel   : conv3x3 (16->32) + bias + relu, implicit GEMM,
//                         K=144->160, N=32 (2 tiles), 10 WMMAs/wave/row.
//                         Input staging uses GLOBAL_LOAD_ASYNC_TO_LDS_B32
//                         (ASYNCcnt) when the toolchain exposes the builtin.
//
// Fragment construction (branchless, no exec divergence):
//   A (pixels):  per-lane k = k0(compile-time) + khi(lane in {0,8});
//                both LDS offsets folded at compile time, selected with
//                v_cndmask; pad channels in the LDS tile hold zeros.
//   B (weights): k-innermost LDS layout -> two ds_load_b128 per fragment.
// ---------------------------------------------------------------------------

typedef __attribute__((ext_vector_type(16))) _Float16 v16h;
typedef __attribute__((ext_vector_type(8)))  _Float16 v8h;
typedef __attribute__((ext_vector_type(8)))  float    v8f;
typedef __attribute__((ext_vector_type(4)))  float    v4f;

#define IMG_H 512
#define IMG_W 512
#define IMG_HW (IMG_H * IMG_W)
#define NSTEP 64            // B*T = 4*16 sequential LIF steps
#define XSTR  68            // xs tile row stride (halves); body at col 2 -> 4B
                            // aligned LDS addresses for async b32 copies

// CDNA5 async global->LDS copy (tracked by ASYNCcnt), when available.
#if defined(__HIP_DEVICE_COMPILE__) && \
    __has_builtin(__builtin_amdgcn_global_load_async_to_lds_b32)
#define HAVE_ASYNC_LDS 1
__device__ __forceinline__ void async_copy_b32(const _Float16* g, _Float16* l) {
    __builtin_amdgcn_global_load_async_to_lds_b32(
        (__attribute__((address_space(1))) int*)(g),
        (__attribute__((address_space(3))) int*)(l), 0, 0);
}
__device__ __forceinline__ void wait_asynccnt0() {
#if __has_builtin(__builtin_amdgcn_s_wait_asynccnt)
    __builtin_amdgcn_s_wait_asynccnt(0);
#else
    asm volatile("s_wait_asynccnt 0" ::: "memory");
#endif
}
#endif

// im2col LDS offset for row k: tile layout [c][dy][XSTR], k = c*9 + dy*3 + dx.
// Tap column for pixel p, offset dx lives at index aoff(k) + 1 + p.
__host__ __device__ constexpr int aoff(int k) {
    return ((k / 9) * 3 + (k % 9) / 3) * XSTR + (k % 9) % 3;
}

// ===========================================================================
// Kernel 1: fused LIF + encoder convs + mean (one block = 32x32 output tile)
// ===========================================================================
__global__ __launch_bounds__(256)
void enc_lif_conv_kernel(const float* __restrict__ ev_on,
                         const float* __restrict__ ev_off,
                         const float* __restrict__ on_w1,
                         const float* __restrict__ on_w2,
                         const float* __restrict__ off_w1,
                         const float* __restrict__ off_w2,
                         float* __restrict__ mean_on,
                         float* __restrict__ mean_off)
{
    const int enc = blockIdx.z;
    const float* __restrict__ ev  = enc ? ev_off : ev_on;
    const float* __restrict__ w1g = enc ? off_w1 : on_w1;
    const float* __restrict__ w2g = enc ? off_w2 : on_w2;
    float* __restrict__ outm      = enc ? mean_off : mean_on;

    __shared__ _Float16 s_tile[36 * 36];     // spikes, halo 2
    __shared__ float    y1[4][34 * 34];      // conv1 output, halo 1

    float w1[4][9], w2[4][9];
#pragma unroll
    for (int c = 0; c < 4; ++c)
#pragma unroll
        for (int k = 0; k < 9; ++k) {
            w1[c][k] = w1g[c * 9 + k];
            w2[c][k] = w2g[c * 9 + k];
        }

    const int tid = threadIdx.x;
    const int y0  = blockIdx.y * 32;
    const int x0  = blockIdx.x * 32;

    // Persistent LIF membrane state: up to 6 halo pixels per thread.
    float v[6];
    int   goff[6];
    bool  inb[6];
#pragma unroll
    for (int i = 0; i < 6; ++i) {
        v[i] = 0.f; inb[i] = false; goff[i] = 0;
        const int idx = tid + i * 256;
        if (idx < 36 * 36) {
            const int r  = idx / 36, cc = idx % 36;
            const int gy = y0 - 2 + r, gx = x0 - 2 + cc;
            inb[i]  = (gy >= 0) && (gy < IMG_H) && (gx >= 0) && (gx < IMG_W);
            goff[i] = gy * IMG_W + gx;
        }
    }

    const int oy = tid >> 3;          // 0..31
    const int ox = (tid & 7) * 4;     // 0,4,...,28
    float o[4] = {0.f, 0.f, 0.f, 0.f};

    for (int t = 0; t < NSTEP; ++t) {
        // -------- LIF charge / fire / hard-reset; spikes -> LDS ----------
#pragma unroll
        for (int i = 0; i < 6; ++i) {
            const int idx = tid + i * 256;
            if (idx < 36 * 36) {
                float s = 0.f;
                if (inb[i]) {
                    const float x = ev[(size_t)t * IMG_HW + goff[i]];
                    if (i == 0 && t + 1 < NSTEP)
                        __builtin_prefetch(&ev[(size_t)(t + 1) * IMG_HW + goff[i]], 0, 1);
                    const float vv = v[i] + (x - v[i]) * 0.1f;   // tau = 10
                    s    = (vv - 1.0f >= 0.f) ? 1.f : 0.f;       // v_th = 1
                    v[i] = vv * (1.f - s);                       // hard reset
                }
                s_tile[idx] = (_Float16)s;
            }
        }
        __syncthreads();

        // -------- conv1 (1->4) + relu on 34x34 ---------------------------
#pragma unroll
        for (int i = 0; i < 5; ++i) {
            const int pos = tid + i * 256;
            if (pos < 34 * 34) {
                const int yy = pos / 34, xx = pos % 34;
                float sv[9];
#pragma unroll
                for (int dy = 0; dy < 3; ++dy)
#pragma unroll
                    for (int dx = 0; dx < 3; ++dx)
                        sv[dy * 3 + dx] = (float)s_tile[(yy + dy) * 36 + xx + dx];
#pragma unroll
                for (int c = 0; c < 4; ++c) {
                    float a = 0.f;
#pragma unroll
                    for (int k = 0; k < 9; ++k) a += w1[c][k] * sv[k];
                    y1[c][pos] = a > 0.f ? a : 0.f;
                }
            }
        }
        __syncthreads();

        // -------- conv2 (4->1) + relu, accumulate temporal mean ----------
        // (no trailing barrier needed: next-iter writes are ordered by the
        //  two collective barriers above)
        float t2[4] = {0.f, 0.f, 0.f, 0.f};
#pragma unroll
        for (int c = 0; c < 4; ++c)
#pragma unroll
            for (int dy = 0; dy < 3; ++dy) {
                float r[6];
#pragma unroll
                for (int j = 0; j < 6; ++j)
                    r[j] = y1[c][(oy + dy) * 34 + ox + j];
#pragma unroll
                for (int p = 0; p < 4; ++p)
#pragma unroll
                    for (int dx = 0; dx < 3; ++dx)
                        t2[p] += w2[c][dy * 3 + dx] * r[p + dx];
            }
#pragma unroll
        for (int p = 0; p < 4; ++p) o[p] += (t2[p] > 0.f ? t2[p] : 0.f);
    }

    const v4f ov = {o[0] * (1.f / 64.f), o[1] * (1.f / 64.f),
                    o[2] * (1.f / 64.f), o[3] * (1.f / 64.f)};
    *(v4f*)&outm[(size_t)(y0 + oy) * IMG_W + (x0 + ox)] = ov;
}

// ===========================================================================
// Kernel 2: clip conv1 (5->16) via WMMA. 128 thr (4 waves) = 64 px, 4 rows.
// ===========================================================================
#define C_PAD1   8      // pad channels so every compile-time offset is valid
#define WSTR1    72     // weight row stride in halves (16B-aligned, K_PAD=64)

__global__ __launch_bounds__(128)
void clip1_wmma_kernel(const float* __restrict__ mean_on,
                       const float* __restrict__ mean_off,
                       const float* __restrict__ rgb,
                       const float* __restrict__ w1g,   // [16][5][3][3]
                       const float* __restrict__ b1g,   // [16]
                       _Float16* __restrict__ y16)      // [16][H][W] f16
{
    __shared__ _Float16 xs[C_PAD1 * 3 * XSTR]; // [c][dy][col'], pad ch = 0
    __shared__ _Float16 ws[16 * WSTR1];        // [n][k], K 45 -> 64 zero-pad
    __shared__ _Float16 ostg[16 * 64];         // D transpose for coalescing

    const int tid    = threadIdx.x;
    const int x_base = blockIdx.x * 64;

    // ---- stage weights once per block (k-innermost, zero-padded) ----
    {
        const int n = tid >> 3;       // 0..15
        const int q = tid & 7;        // 0..7
#pragma unroll
        for (int i = 0; i < 9; ++i) { // 8*9 = 72 = WSTR1
            const int k = q * 9 + i;
            ws[n * WSTR1 + k] = (_Float16)((k < 45) ? w1g[n * 45 + k] : 0.f);
        }
    }

    const int lane = tid & 31;
    const int wave = tid >> 5;
    const bool hi  = (lane & 16) != 0;
    const int n    = lane & 15;
    const int acol = wave * 16 + (lane & 15) + 1;   // +1: body starts at col'=1
    const float bias = b1g[n];

    for (int ry = 0; ry < 4; ++ry) {
        const int y = blockIdx.y * 4 + ry;

        // ---- stage input tile: col' in [1,66] maps to gx = x_base+col'-2 ----
        for (int idx = tid; idx < C_PAD1 * 3 * XSTR; idx += 128) {
            const int c    = idx / (3 * XSTR);
            const int rem  = idx - c * (3 * XSTR);
            const int dy   = rem / XSTR;
            const int colp = rem - dy * XSTR;
            float val = 0.f;
            if (c < 5 && colp >= 1 && colp <= 66) {
                const int gy = y + dy - 1;
                const int gx = x_base + colp - 2;
                if (gy >= 0 && gy < IMG_H && gx >= 0 && gx < IMG_W) {
                    const size_t p = (size_t)gy * IMG_W + gx;
                    val = (c == 0) ? mean_on[p]
                        : (c == 1) ? mean_off[p]
                                   : rgb[(size_t)(c - 2) * IMG_HW + p];
                }
            }
            xs[idx] = (_Float16)val;
        }
        __syncthreads();

        // ---- WMMA: K = 64 (2 k-steps) ----
        v8f acc = {};
#pragma unroll
        for (int ks = 0; ks < 2; ++ks) {
            v16h a;
#pragma unroll
            for (int e = 0; e < 16; ++e) {
                const int k0 = ((e < 8) ? e : e + 8) + ks * 32; // compile-time
                const int off = hi ? aoff(k0 + 8) : aoff(k0);   // v_cndmask
                a[e] = xs[off + acol];
            }
            const int kb = ks * 32 + (hi ? 8 : 0);
            const v8h blo = *(const v8h*)&ws[n * WSTR1 + kb];
            const v8h bhi = *(const v8h*)&ws[n * WSTR1 + kb + 16];
            const v16h b = __builtin_shufflevector(blo, bhi,
                0, 1, 2, 3, 4, 5, 6, 7, 8, 9, 10, 11, 12, 13, 14, 15);
            acc = __builtin_amdgcn_wmma_f32_16x16x32_f16(
                      false, a, false, b, (short)0, acc, false, false);
        }

        // ---- epilogue: bias + relu -> LDS transpose -> coalesced stores ----
#pragma unroll
        for (int r = 0; r < 8; ++r) {
            const int mm = r + (hi ? 8 : 0);
            float val = acc[r] + bias;
            val = val > 0.f ? val : 0.f;
            ostg[n * 64 + wave * 16 + mm] = (_Float16)val;
        }
        __syncthreads();
        {
            const int n2  = tid >> 3;   // 0..15
            const int sub = tid & 7;    // 0..7  (8 halves = 16B each)
            const v8h vals = *(const v8h*)&ostg[n2 * 64 + sub * 8];
            *(v8h*)&y16[(size_t)n2 * IMG_HW + (size_t)y * IMG_W
                        + x_base + sub * 8] = vals;
        }
        // no trailing barrier: next-iter xs writes can only start after every
        // wave passed the post-WMMA barrier above
    }
}

// ===========================================================================
// Kernel 3: clip conv2 (16->32) via WMMA. K=144->160, two N tiles, 4 rows.
// ===========================================================================
#define C_PAD2   18     // covers aoff(k) up to k=161 (max k used is 159)
#define WSTR2    168    // weight row stride in halves (16B-aligned, K_PAD=160)

__global__ __launch_bounds__(128)
void clip2_wmma_kernel(const _Float16* __restrict__ y16,  // [16][H][W]
                       const float* __restrict__ w2g,     // [32][16][3][3]
                       const float* __restrict__ b2g,     // [32]
                       float* __restrict__ out)           // [32][H][W]
{
    __shared__ _Float16 xs[C_PAD2 * 3 * XSTR];
    __shared__ _Float16 ws[32 * WSTR2];
    __shared__ float    ostg[32 * 64];

    const int tid    = threadIdx.x;
    const int x_base = blockIdx.x * 64;

    // ---- stage weights once per block ----
    {
        const int n = tid >> 2;        // 0..31
        const int q = tid & 3;         // 0..3
#pragma unroll
        for (int i = 0; i < 42; ++i) { // 4*42 = 168 = WSTR2
            const int k = q * 42 + i;
            ws[n * WSTR2 + k] = (_Float16)((k < 144) ? w2g[n * 144 + k] : 0.f);
        }
    }

    const int lane = tid & 31;
    const int wave = tid >> 5;
    const bool hi  = (lane & 16) != 0;
    const int n    = lane & 15;
    const int acol = wave * 16 + (lane & 15) + 1;
    const float bias0 = b2g[n], bias1 = b2g[16 + n];

#if HAVE_ASYNC_LDS
    // zero-fill pad channels (c = 16,17) once; never touched again
    for (int idx = tid; idx < 2 * 3 * XSTR; idx += 128)
        xs[16 * 3 * XSTR + idx] = (_Float16)0.f;
#endif

    for (int ry = 0; ry < 4; ++ry) {
        const int y = blockIdx.y * 4 + ry;

#if HAVE_ASYNC_LDS
        // ---- body: async global->LDS b32 copies (cols 0..63 -> col' 2..65) ----
        // 48 rows (16 ch x 3 dy); rowi = i*4 + wave is wave-uniform.
#pragma unroll
        for (int i = 0; i < 12; ++i) {
            const int rowi = i * 4 + wave;          // 0..47
            const int c    = rowi / 3;
            const int dy   = rowi - 3 * c;
            const int gy   = y + dy - 1;
            _Float16* ldst = &xs[(c * 3 + dy) * XSTR + 2 + 2 * lane];
            if (gy >= 0 && gy < IMG_H) {
                async_copy_b32(&y16[(size_t)c * IMG_HW + (size_t)gy * IMG_W
                                    + x_base + 2 * lane], ldst);
            } else {
                *(int*)ldst = 0;                    // OOB row -> zeros
            }
        }
        // ---- borders: col' = 1 (gx = x_base-1) and col' = 66 (gx = x_base+64)
        if (tid < 96) {
            const int rowi = tid >> 1;              // 0..47
            const int side = tid & 1;
            const int c    = rowi / 3;
            const int dy   = rowi - 3 * c;
            const int colp = side ? 66 : 1;
            const int gy   = y + dy - 1;
            const int gx   = x_base + colp - 2;
            _Float16 val = (_Float16)0.f;
            if (gy >= 0 && gy < IMG_H && gx >= 0 && gx < IMG_W)
                val = y16[(size_t)c * IMG_HW + (size_t)gy * IMG_W + gx];
            xs[(c * 3 + dy) * XSTR + colp] = val;
        }
        wait_asynccnt0();
#else
        // ---- synchronous fallback staging ----
        for (int idx = tid; idx < C_PAD2 * 3 * XSTR; idx += 128) {
            const int c    = idx / (3 * XSTR);
            const int rem  = idx - c * (3 * XSTR);
            const int dy   = rem / XSTR;
            const int colp = rem - dy * XSTR;
            _Float16 val = (_Float16)0.f;
            if (c < 16 && colp >= 1 && colp <= 66) {
                const int gy = y + dy - 1;
                const int gx = x_base + colp - 2;
                if (gy >= 0 && gy < IMG_H && gx >= 0 && gx < IMG_W)
                    val = y16[(size_t)c * IMG_HW + (size_t)gy * IMG_W + gx];
            }
            xs[idx] = val;
        }
#endif
        __syncthreads();

        // ---- WMMA: K = 160 (5 k-steps) x 2 N tiles ----
        v8f acc0 = {}, acc1 = {};
#pragma unroll
        for (int ks = 0; ks < 5; ++ks) {
            v16h a;
#pragma unroll
            for (int e = 0; e < 16; ++e) {
                const int k0 = ((e < 8) ? e : e + 8) + ks * 32;
                const int off = hi ? aoff(k0 + 8) : aoff(k0);
                a[e] = xs[off + acol];
            }
            const int kb = ks * 32 + (hi ? 8 : 0);
            const v8h b0lo = *(const v8h*)&ws[n * WSTR2 + kb];
            const v8h b0hi = *(const v8h*)&ws[n * WSTR2 + kb + 16];
            const v8h b1lo = *(const v8h*)&ws[(16 + n) * WSTR2 + kb];
            const v8h b1hi = *(const v8h*)&ws[(16 + n) * WSTR2 + kb + 16];
            const v16h b0 = __builtin_shufflevector(b0lo, b0hi,
                0, 1, 2, 3, 4, 5, 6, 7, 8, 9, 10, 11, 12, 13, 14, 15);
            const v16h b1 = __builtin_shufflevector(b1lo, b1hi,
                0, 1, 2, 3, 4, 5, 6, 7, 8, 9, 10, 11, 12, 13, 14, 15);
            acc0 = __builtin_amdgcn_wmma_f32_16x16x32_f16(
                       false, a, false, b0, (short)0, acc0, false, false);
            acc1 = __builtin_amdgcn_wmma_f32_16x16x32_f16(
                       false, a, false, b1, (short)0, acc1, false, false);
        }

        // ---- epilogue: bias + relu -> LDS transpose -> coalesced stores ----
#pragma unroll
        for (int r = 0; r < 8; ++r) {
            const int mm = r + (hi ? 8 : 0);
            float v0 = acc0[r] + bias0; v0 = v0 > 0.f ? v0 : 0.f;
            float v1 = acc1[r] + bias1; v1 = v1 > 0.f ? v1 : 0.f;
            ostg[n * 64 + wave * 16 + mm]        = v0;
            ostg[(16 + n) * 64 + wave * 16 + mm] = v1;
        }
        __syncthreads();
        {
            const int n2  = tid >> 2;   // 0..31
            const int sub = tid & 3;    // 0..3  (16 floats each)
#pragma unroll
            for (int j = 0; j < 4; ++j) {
                const v4f vals = *(const v4f*)&ostg[n2 * 64 + sub * 16 + j * 4];
                *(v4f*)&out[(size_t)n2 * IMG_HW + (size_t)y * IMG_W
                            + x_base + sub * 16 + j * 4] = vals;
            }
        }
        // no trailing barrier (see clip1)
    }
}

// ===========================================================================
extern "C" void kernel_launch(void* const* d_in, const int* in_sizes, int n_in,
                              void* d_out, int out_size, void* d_ws, size_t ws_size,
                              hipStream_t stream)
{
    const float* On      = (const float*)d_in[0];
    const float* Off     = (const float*)d_in[1];
    const float* rgb     = (const float*)d_in[2];
    const float* on_w1   = (const float*)d_in[3];
    const float* on_w2   = (const float*)d_in[4];
    const float* off_w1  = (const float*)d_in[5];
    const float* off_w2  = (const float*)d_in[6];
    const float* clip_w1 = (const float*)d_in[7];
    const float* clip_b1 = (const float*)d_in[8];
    const float* clip_w2 = (const float*)d_in[9];
    const float* clip_b2 = (const float*)d_in[10];
    float* out = (float*)d_out;

    char* ws = (char*)d_ws;
    float*    mean_on  = (float*)ws;                                    // 1 MB
    float*    mean_off = (float*)(ws + (size_t)IMG_HW * sizeof(float)); // 1 MB
    _Float16* y16 = (_Float16*)(ws + (size_t)2 * IMG_HW * sizeof(float)); // 8.4 MB

    hipLaunchKernelGGL(enc_lif_conv_kernel, dim3(16, 16, 2), dim3(256), 0, stream,
                       On, Off, on_w1, on_w2, off_w1, off_w2, mean_on, mean_off);
    hipLaunchKernelGGL(clip1_wmma_kernel, dim3(8, 128), dim3(128), 0, stream,
                       mean_on, mean_off, rgb, clip_w1, clip_b1, y16);
    hipLaunchKernelGGL(clip2_wmma_kernel, dim3(8, 128), dim3(128), 0, stream,
                       y16, clip_w2, clip_b2, out);
}
